// ConvAttnFauxLarsen_14130442404233
// MI455X (gfx1250) — matrix operand order repaired and checked by reading, not compiled, not run on hardware
//
#include <hip/hip_runtime.h>
#include <math.h>

#define C    128
#define KT   7
#define BB   64
#define MM   4
#define DD   16
#define TLEN 202
#define SEQLEN 194

typedef __attribute__((ext_vector_type(16))) __bf16 v16bf;
typedef __attribute__((ext_vector_type(8)))  __bf16 v8bf;
typedef __attribute__((ext_vector_type(8)))  float  v8f;

// Branchless tanh: prefer gfx1250 hardware v_tanh_f32.
static __device__ __forceinline__ float fast_tanh(float x) {
#if __has_builtin(__builtin_amdgcn_tanhf)
    return __builtin_amdgcn_tanhf(x);
#else
    float ax = __builtin_fabsf(x);
    float e  = __builtin_amdgcn_exp2f(ax * -2.8853900817779268f);
    float r  = __builtin_amdgcn_rcpf(1.f + e);
    float t  = (1.f - e) * r;
    return x >= 0.f ? t : -t;
#endif
}

// Load one 32x16 bf16 B fragment set (4 K-slabs of 32) for column c.
static __device__ __forceinline__ void load_bfrags(v16bf f[4], const __bf16* Wbase,
                                                   int c, int hi) {
#pragma unroll
    for (int kk = 0; kk < 4; ++kk) {
        v8bf lo = *(const v8bf*)(Wbase + (size_t)c * C + kk * 32 + hi * 8);
        v8bf hv = *(const v8bf*)(Wbase + (size_t)c * C + kk * 32 + hi * 8 + 16);
        f[kk] = __builtin_shufflevector(lo, hv, 0, 1, 2, 3, 4, 5, 6, 7,
                                        8, 9, 10, 11, 12, 13, 14, 15);
    }
}

// Build a 16x32 bf16 A fragment from a bf16 LDS row (two 16B ds loads + concat).
static __device__ __forceinline__ v16bf make_afrag(const __bf16* rowp, int kb) {
    v8bf lo = *(const v8bf*)(rowp + kb);
    v8bf hv = *(const v8bf*)(rowp + kb + 16);
    return __builtin_shufflevector(lo, hv, 0, 1, 2, 3, 4, 5, 6, 7,
                                           8, 9, 10, 11, 12, 13, 14, 15);
}

// ---------------------------------------------------------------------------
// Weight prep: blk_weights (M,D,C,C,K) f32 -> w1t bf16 [mi][k][c_out][c_in]
// ---------------------------------------------------------------------------
__global__ void k_prep_w1(const float* __restrict__ blk, __bf16* __restrict__ w1t) {
    size_t idx = (size_t)blockIdx.x * 256 + threadIdx.x;
    size_t total = (size_t)MM * DD * KT * C * C;
    if (idx >= total) return;
    size_t r = idx;
    int cp = (int)(r % C); r /= C;   // c_in
    int d  = (int)(r % C); r /= C;   // c_out
    int k  = (int)(r % KT); r /= KT;
    int mi = (int)r;
    w1t[idx] = (__bf16)blk[(((size_t)mi * C + d) * C + cp) * KT + k];
}

// pw_w (M,D,C,C) f32 [ci][co] -> pwt bf16 [mi][co][ci]
__global__ void k_prep_pw(const float* __restrict__ pw, __bf16* __restrict__ pwt) {
    size_t idx = (size_t)blockIdx.x * 256 + threadIdx.x;
    size_t total = (size_t)MM * DD * C * C;
    if (idx >= total) return;
    size_t r = idx;
    int ci = (int)(r % C); r /= C;
    int co = (int)(r % C); r /= C;
    int mi = (int)r;
    pwt[idx] = (__bf16)pw[((size_t)mi * C + ci) * C + co];
}

// ---------------------------------------------------------------------------
// Input projection: z[b,t,c] = seq[b,t] * in_w[c] + in_b[c]
// ---------------------------------------------------------------------------
__global__ void k_inproj(const float* __restrict__ seq, int sstride,
                         const float* __restrict__ inw, const float* __restrict__ inb,
                         float* __restrict__ z) {
    int idx = blockIdx.x * 256 + threadIdx.x;
    if (idx >= BB * SEQLEN * C) return;
    int c = idx & (C - 1);
    int bt = idx >> 7;
    int t = bt % SEQLEN;
    int b = bt / SEQLEN;
    z[idx] = seq[(size_t)b * sstride + t] * inw[c] + inb[c];
}

// ---------------------------------------------------------------------------
// Fused conv block. One workgroup = (batch b) x (16 output positions).
// 4 wave32s; each wave owns 2 column tiles of 16 channels.
// ---------------------------------------------------------------------------
__global__ __launch_bounds__(128, 4) void k_convblock(
    const float* __restrict__ zin, int Lin, int Lo,
    int s, int J, int wOff,
    const __bf16* __restrict__ W1t,   // [KT][C][C]  (c_out, c_in)
    const __bf16* __restrict__ pwT,   // [C][C]      (c_out, c_in)
    const float* __restrict__ ls, const float* __restrict__ lb,
    const float* __restrict__ pwb, const float* __restrict__ pa_p,
    float* __restrict__ zout)
{
    __shared__ float  zs[44][C + 1];     // f32 rows (FIR taps + residual)
    __shared__ __bf16 zbf[44][C + 8];    // bf16 rows (A fragments), 272B stride
    __shared__ float  yt[16][C + 1];
    __shared__ float  red1[16][8], red2[16][8];
    __shared__ float  lnm[16], lnr[16];

    const int tid   = threadIdx.x;
    const int lane  = tid & 31;
    const int wave  = tid >> 5;
    const int row16 = lane & 15;
    const int hi    = lane >> 4;
    const int b     = blockIdx.y;
    const int l0    = blockIdx.x * 16;
    const float pa  = pa_p[0];

    // Stage activation rows [s*l0 .. s*l0 + s*21 + J - 1] (clamped) into LDS.
    const int nstage = s * 21 + J;
    const int g0 = s * l0;
    for (int idx = tid; idx < nstage * C; idx += 128) {
        int r = idx >> 7, c = idx & (C - 1);
        int gr = g0 + r; if (gr > Lin - 1) gr = Lin - 1;
        float v = zin[((size_t)b * Lin + gr) * C + c];
        zs[r][c]  = v;
        zbf[r][c] = (__bf16)v;
    }
    __syncthreads();

    const int c_t0 = (wave * 2) * 16 + row16;
    const int c_t1 = (wave * 2 + 1) * 16 + row16;

    float acc0[8], acc1[8];
#pragma unroll
    for (int r = 0; r < 8; ++r) { acc0[r] = 0.f; acc1[r] = 0.f; }

    for (int j = 0; j < J; ++j) {
        // A fragments via two 16B LDS loads each (bf16 rows, WMMA 16x32 layout).
        v16bf a[4];
        const __bf16* rowp = &zbf[s * row16 + wOff + j][0];
#pragma unroll
        for (int kk = 0; kk < 4; ++kk)
            a[kk] = make_afrag(rowp, kk * 32 + hi * 8);

#pragma unroll 1
        for (int k = 0; k < KT; ++k) {
            const __bf16* Wk = W1t + (size_t)k * C * C;
            v16bf b0[4], b1[4];
            load_bfrags(b0, Wk, c_t0, hi);
            load_bfrags(b1, Wk, c_t1, hi);
            v8f g0 = {}, g1 = {};
#pragma unroll
            for (int kk = 0; kk < 4; ++kk)
                g0 = __builtin_amdgcn_wmma_f32_16x16x32_bf16(
                        false, a[kk], false, b0[kk], (short)0, g0, false, false);
#pragma unroll
            for (int kk = 0; kk < 4; ++kk)
                g1 = __builtin_amdgcn_wmma_f32_16x16x32_bf16(
                        false, a[kk], false, b1[kk], (short)0, g1, false, false);
#pragma unroll
            for (int r = 0; r < 8; ++r) {
                int Mr = hi * 8 + r;
                int pr = s * Mr + s * k + j;
                acc0[r] += fast_tanh(g0[r]) * zs[pr][c_t0];
                acc1[r] += fast_tanh(g1[r]) * zs[pr][c_t1];
            }
        }
    }

    // residual + write y tile
#pragma unroll
    for (int r = 0; r < 8; ++r) {
        int Mr = hi * 8 + r;
        int rr = s * Mr + wOff;
        yt[Mr][c_t0] = zs[rr][c_t0] + acc0[r];
        yt[Mr][c_t1] = zs[rr][c_t1] + acc1[r];
    }
    __syncthreads();

    // layernorm statistics (mean / rsqrt(var+eps) per row)
    {
        int r = tid >> 3, seg = tid & 7;
        float sm = 0.f, sq = 0.f;
#pragma unroll
        for (int e = 0; e < 16; ++e) {
            float v = yt[r][seg * 16 + e];
            sm += v; sq += v * v;
        }
        red1[r][seg] = sm; red2[r][seg] = sq;
    }
    __syncthreads();
    if (tid < 16) {
        float sm = 0.f, sq = 0.f;
#pragma unroll
        for (int e = 0; e < 8; ++e) { sm += red1[tid][e]; sq += red2[tid][e]; }
        float mean = sm * (1.f / C);
        float var  = sq * (1.f / C) - mean * mean;
        lnm[tid] = mean;
        lnr[tid] = rsqrtf(var + 1e-6f);
    }
    __syncthreads();

    // A fragments of normalized y for the pointwise GEMM
    v16bf a2[4];
    {
        const float mean = lnm[row16], rstd = lnr[row16];
#pragma unroll
        for (int kk = 0; kk < 4; ++kk) {
            const int kb = kk * 32 + hi * 8;
#pragma unroll
            for (int e = 0; e < 8; ++e) {
                int cq0 = kb + e, cq1 = kb + 16 + e;
                a2[kk][e]     = (__bf16)((yt[row16][cq0] - mean) * rstd * ls[cq0] + lb[cq0]);
                a2[kk][e + 8] = (__bf16)((yt[row16][cq1] - mean) * rstd * ls[cq1] + lb[cq1]);
            }
        }
    }
    // Preload both tiles' B fragments, then run the two WMMA chains.
    v16bf pf0[4], pf1[4];
    load_bfrags(pf0, pwT, c_t0, hi);
    load_bfrags(pf1, pwT, c_t1, hi);
#pragma unroll
    for (int t = 0; t < 2; ++t) {
        const int c = (t == 0) ? c_t0 : c_t1;
        v8f g = {};
#pragma unroll
        for (int kk = 0; kk < 4; ++kk)
            g = __builtin_amdgcn_wmma_f32_16x16x32_bf16(
                    false, a2[kk], false, (t == 0) ? pf0[kk] : pf1[kk],
                    (short)0, g, false, false);
        const float lsc = ls[c], lbc = lb[c], pb = pwb[c];
#pragma unroll
        for (int r = 0; r < 8; ++r) {
            int Mr = hi * 8 + r;
            int l  = l0 + Mr;
            if (l < Lo) {
                float tv = g[r] + pb;
                float pv = tv >= 0.f ? tv : pa * tv;
                float yn = (yt[Mr][c] - lnm[Mr]) * lnr[Mr] * lsc + lbc;
                zout[((size_t)b * Lo + l) * C + c] = yn + pv;
            }
        }
    }
}

// ---------------------------------------------------------------------------
// Output projection + sliding-window shift.
// ---------------------------------------------------------------------------
__global__ void k_outshift(const float* __restrict__ zfin,
                           const float* __restrict__ outw,
                           const float* __restrict__ outb_p,
                           const float* __restrict__ opa_p,
                           const float* __restrict__ srcseq, int sstride,
                           const float* __restrict__ x, int xoff,
                           float* __restrict__ dstseq)
{
    __shared__ float sd[128];
    __shared__ float sval;
    const int b = blockIdx.x, tid = threadIdx.x;
    if (tid < 128) sd[tid] = zfin[(size_t)b * C + tid] * outw[tid];
    __syncthreads();
    if (tid == 0) {
        float sm = 0.f;
        for (int i = 0; i < 128; ++i) sm += sd[i];
        sm += outb_p[0];
        float opa = opa_p[0];
        sval = sm >= 0.f ? sm : opa * sm;
    }
    __syncthreads();
    for (int t = tid; t < SEQLEN; t += blockDim.x) {
        float v;
        if (t < SEQLEN - 2)       v = srcseq[(size_t)b * sstride + t + 2];
        else if (t == SEQLEN - 2) v = x[(size_t)b * TLEN + xoff];
        else                      v = sval;
        dstseq[(size_t)b * SEQLEN + t] = v;
    }
}

// ---------------------------------------------------------------------------
extern "C" void kernel_launch(void* const* d_in, const int* in_sizes, int n_in,
                              void* d_out, int out_size, void* d_ws, size_t ws_size,
                              hipStream_t stream) {
    (void)in_sizes; (void)n_in; (void)out_size; (void)ws_size;
    const float* x    = (const float*)d_in[0];
    const float* in_w = (const float*)d_in[1];
    const float* in_b = (const float*)d_in[2];
    const float* blkw = (const float*)d_in[3];
    const float* lns  = (const float*)d_in[4];
    const float* lnb  = (const float*)d_in[5];
    const float* pww  = (const float*)d_in[6];
    const float* pwb  = (const float*)d_in[7];
    const float* pra  = (const float*)d_in[8];
    const float* outw = (const float*)d_in[9];
    const float* outb = (const float*)d_in[10];
    const float* opra = (const float*)d_in[11];
    float* out = (float*)d_out;

    char* ws = (char*)d_ws;
    size_t off = 0;
    __bf16* w1t = (__bf16*)(ws + off); off += (size_t)MM * DD * KT * C * C * 2;
    __bf16* pwt = (__bf16*)(ws + off); off += (size_t)MM * DD * C * C * 2;
    off = (off + 255) & ~(size_t)255;
    float* zA   = (float*)(ws + off); off += (size_t)BB * SEQLEN * C * 4;
    float* zB   = (float*)(ws + off); off += (size_t)BB * SEQLEN * C * 4;
    float* seq0 = (float*)(ws + off); off += (size_t)BB * SEQLEN * 4;
    float* seq1 = (float*)(ws + off); off += (size_t)BB * SEQLEN * 4;

    // weight conversion / transposition (bf16)
    {
        size_t tot = (size_t)MM * DD * KT * C * C;
        k_prep_w1<<<dim3((unsigned)((tot + 255) / 256)), 256, 0, stream>>>(blkw, w1t);
        tot = (size_t)MM * DD * C * C;
        k_prep_pw<<<dim3((unsigned)((tot + 255) / 256)), 256, 0, stream>>>(pww, pwt);
    }

    const float* seqsrc = x;
    int sstride = TLEN;
    for (int m = 0; m < MM; ++m) {
        int tot = BB * SEQLEN * C;
        k_inproj<<<dim3((tot + 255) / 256), 256, 0, stream>>>(
            seqsrc, sstride, in_w + m * C, in_b + m * C, zA);

        float* zin = zA;
        float* zo  = zB;
        int Lin = SEQLEN;
        for (int i = 0; i < DD; ++i) {
            int mi = m * DD + i;
            int s_, J_, wOff_, Lo;
            if (i == 0) { s_ = 2; J_ = 2; wOff_ = 12; Lo = (Lin - 2 * KT) / 2 + 1; }
            else        { s_ = 1; J_ = 1; wOff_ = 6;  Lo = Lin - (KT - 1); }
            dim3 grid((Lo + 15) / 16, BB);
            k_convblock<<<grid, 128, 0, stream>>>(
                zin, Lin, Lo, s_, J_, wOff_,
                w1t + (size_t)mi * KT * C * C,
                pwt + (size_t)mi * C * C,
                lns + (size_t)mi * C, lnb + (size_t)mi * C,
                pwb + (size_t)mi * C, pra + mi, zo);
            Lin = Lo;
            float* tmp = zin; zin = zo; zo = tmp;
        }
        float* dst = (m == MM - 1) ? out : ((m & 1) ? seq1 : seq0);
        k_outshift<<<dim3(BB), 256, 0, stream>>>(
            zin, outw + m * C, outb + m, opra + m,
            seqsrc, sstride, x, SEQLEN + 2 * m, dst);
        seqsrc = dst;
        sstride = SEQLEN;
    }
}
